// MiMoMoeDecoderLayer_3728031613185
// MI455X (gfx1250) — compile-verified, matrix operand order checked by reading, not run on hardware
//
#include <hip/hip_runtime.h>
#include <hip/hip_bf16.h>
#include <math.h>

// ---------------- problem constants (match reference) ----------------
#define TT    1024      // tokens
#define HH    2048      // hidden
#define NHQ   16        // q heads
#define NKVH  4         // kv heads
#define DH    128       // head dim
#define NE    16        // experts
#define TOPK  2
#define II    1024      // moe intermediate
#define NSLOT (TT*TOPK) // total (token, expert) slots = 2048

typedef __attribute__((ext_vector_type(16))) _Float16 v16h;
typedef __attribute__((ext_vector_type(8)))  _Float16 v8h;
typedef __attribute__((ext_vector_type(4)))  _Float16 v4h;
typedef __attribute__((ext_vector_type(2)))  _Float16 v2h;
typedef __attribute__((ext_vector_type(8)))  float    v8f;

// LDS row stride in halves: 40 halves = 80B (multiple of 16B for b128 loads)
#define LSTRIDE 40

// Build a 16-half WMMA fragment from a fragment-ready LDS row:
// lane-half h reads k = [h*8 .. h*8+7] and [16+h*8 .. 16+h*8+7]
// -> exactly two ds_load_b128.
__device__ __forceinline__ v16h ldfrag(const _Float16* p, int h)
{
    const v8h lo = *(const v8h*)(p + (h << 3));
    const v8h hi = *(const v8h*)(p + 16 + (h << 3));
    return __builtin_shufflevector(lo, hi, 0, 1, 2, 3, 4, 5, 6, 7,
                                           8, 9, 10, 11, 12, 13, 14, 15);
}

// =====================================================================
// RMSNorm: one block per token
// =====================================================================
__global__ __launch_bounds__(256) void rmsnorm_kernel(
    const float* __restrict__ x, const float* __restrict__ w,
    float* __restrict__ o)
{
    __shared__ float red[256];
    const int t = blockIdx.x;
    const float* xr = x + (size_t)t * HH;
    float s = 0.f;
    for (int i = threadIdx.x; i < HH; i += 256) { float v = xr[i]; s += v * v; }
    red[threadIdx.x] = s;
    __syncthreads();
    for (int st = 128; st > 0; st >>= 1) {
        if (threadIdx.x < st) red[threadIdx.x] += red[threadIdx.x + st];
        __syncthreads();
    }
    const float rms = rsqrtf(red[0] / (float)HH + 1e-6f);
    float* orow = o + (size_t)t * HH;
    for (int i = threadIdx.x; i < HH; i += 256) orow[i] = xr[i] * rms * w[i];
}

// =====================================================================
// Generic WMMA GEMM: C[M,N] = A[M,K] @ B[K,N] (+bias) (+addsrc)
// fp32 in/out, f16 WMMA 16x16x32, fp32 accum.
// Block: 256 thr = 8 waves -> 32x128 tile; each wave 16x32 (2 wmma/step).
// =====================================================================
__global__ __launch_bounds__(256) void gemm_wmma_kernel(
    const float* __restrict__ A, const float* __restrict__ B,
    const float* __restrict__ bias, const float* __restrict__ addsrc,
    float* __restrict__ C, int M, int N, int K)
{
    __shared__ _Float16 As[32][LSTRIDE];      // A tile, row-major [m][k]
    __shared__ _Float16 BsT[128][LSTRIDE];    // B tile, transposed [n][k]
    const int tid  = threadIdx.x;
    const int wave = tid >> 5, lane = tid & 31;
    const int tm = (wave >> 2) << 4;          // 0 / 16
    const int tn = (wave & 3) << 5;           // 0 / 32 / 64 / 96
    const int bm = blockIdx.x * 32;
    const int bn = blockIdx.y * 128;
    if (bm >= M) return;
    const int lr = tid >> 3;                  // A: row 0..31
    const int la = (tid & 7) << 2;            // A: 4 cols per thread
    const int kk  = (tid >> 4) << 1;          // B: even k row 0..30
    const int nn0 = (tid & 15) << 3;          // B: 8 cols per thread
    const int hhalf = lane >> 4, mr = lane & 15;
    v8f acc0 = {}, acc1 = {};
    for (int k0 = 0; k0 < K; k0 += 32) {
        // ---- stage A tile (fp32 -> f16), one b64 store ----
        {
            const int row = bm + lr;
            v4h a;
            if (row < M) {
                const float4 f = *(const float4*)(A + (size_t)row * K + k0 + la);
                a[0] = (_Float16)f.x; a[1] = (_Float16)f.y;
                a[2] = (_Float16)f.z; a[3] = (_Float16)f.w;
            } else {
                a[0] = a[1] = a[2] = a[3] = (_Float16)0.f;
            }
            *(v4h*)&As[lr][la] = a;
        }
        // ---- stage B tile transposed: 2 k-rows x 8 n, packed b32 stores ----
        {
            const float* bp0 = B + (size_t)(k0 + kk) * N + bn + nn0;
            const float* bp1 = bp0 + N;
            const float4 r00 = *(const float4*)(bp0);
            const float4 r01 = *(const float4*)(bp0 + 4);
            const float4 r10 = *(const float4*)(bp1);
            const float4 r11 = *(const float4*)(bp1 + 4);
            v2h t0; t0[0] = (_Float16)r00.x; t0[1] = (_Float16)r10.x; *(v2h*)&BsT[nn0+0][kk] = t0;
            v2h t1; t1[0] = (_Float16)r00.y; t1[1] = (_Float16)r10.y; *(v2h*)&BsT[nn0+1][kk] = t1;
            v2h t2; t2[0] = (_Float16)r00.z; t2[1] = (_Float16)r10.z; *(v2h*)&BsT[nn0+2][kk] = t2;
            v2h t3; t3[0] = (_Float16)r00.w; t3[1] = (_Float16)r10.w; *(v2h*)&BsT[nn0+3][kk] = t3;
            v2h t4; t4[0] = (_Float16)r01.x; t4[1] = (_Float16)r11.x; *(v2h*)&BsT[nn0+4][kk] = t4;
            v2h t5; t5[0] = (_Float16)r01.y; t5[1] = (_Float16)r11.y; *(v2h*)&BsT[nn0+5][kk] = t5;
            v2h t6; t6[0] = (_Float16)r01.z; t6[1] = (_Float16)r11.z; *(v2h*)&BsT[nn0+6][kk] = t6;
            v2h t7; t7[0] = (_Float16)r01.w; t7[1] = (_Float16)r11.w; *(v2h*)&BsT[nn0+7][kk] = t7;
        }
        if (k0 + 32 < K)
            __builtin_prefetch(B + (size_t)(k0 + 32 + kk) * N + bn + nn0, 0, 1);
        __syncthreads();
        // ---- fragments: 2 x ds_load_b128 each ----
        const v16h af  = ldfrag(&As[tm + mr][0], hhalf);
        const v16h bf0 = ldfrag(&BsT[tn + mr][0], hhalf);
        const v16h bf1 = ldfrag(&BsT[tn + 16 + mr][0], hhalf);
        acc0 = __builtin_amdgcn_wmma_f32_16x16x32_f16(
                   false, af, false, bf0, (short)0, acc0, false, false);
        acc1 = __builtin_amdgcn_wmma_f32_16x16x32_f16(
                   false, af, false, bf1, (short)0, acc1, false, false);
        __syncthreads();
    }
    const int col0 = bn + tn + mr;
    const int col1 = col0 + 16;
    #pragma unroll
    for (int r = 0; r < 8; ++r) {
        const int row = bm + tm + (hhalf << 3) + r;
        if (row < M) {
            float x0 = acc0[r], x1 = acc1[r];
            if (bias)   { x0 += bias[col0]; x1 += bias[col1]; }
            if (addsrc) { x0 += addsrc[(size_t)row * N + col0];
                          x1 += addsrc[(size_t)row * N + col1]; }
            C[(size_t)row * N + col0] = x0;
            C[(size_t)row * N + col1] = x1;
        }
    }
}

// =====================================================================
// RoPE (neox style), in place over [T, nheads, 128]
// =====================================================================
__global__ __launch_bounds__(256) void rope_kernel(
    float* __restrict__ x, const int* __restrict__ pos, int nheads)
{
    const int gid = blockIdx.x * 256 + threadIdx.x;
    const int total = TT * nheads * (DH / 2);
    if (gid >= total) return;
    const int j  = gid & 63;
    const int hh = (gid >> 6) % nheads;
    const int t  = gid / (64 * nheads);
    const float inv = __powf(1.0e6f, -(float)j / 64.0f);
    const float ang = (float)pos[t] * inv;
    const float c = __cosf(ang), s = __sinf(ang);
    float* p = x + (size_t)t * (nheads * DH) + hh * DH + j;
    const float x1 = p[0], x2 = p[64];
    p[0]  = x1 * c - x2 * s;
    p[64] = x2 * c + x1 * s;
}

// =====================================================================
// Causal flash attention, one wave per (head, query-row); K/V live in L2
// =====================================================================
__global__ __launch_bounds__(256) void attn_kernel(
    const float* __restrict__ q, const float* __restrict__ k,
    const float* __restrict__ v, float* __restrict__ o)
{
    const int wave = threadIdx.x >> 5, lane = threadIdx.x & 31;
    const int wid  = blockIdx.x * 8 + wave;   // 0 .. T*NH-1
    const int head = wid >> 10;               // / T (T = 1024)
    const int t    = wid & (TT - 1);
    const int kvh  = head >> 2;               // NH/NKV = 4
    const float scale = 0.08838834764831845f; // 1/sqrt(128)
    float4 qv = *(const float4*)(q + (size_t)t * (NHQ * DH) + head * DH + lane * 4);
    qv.x *= scale; qv.y *= scale; qv.z *= scale; qv.w *= scale;
    float m = -INFINITY, l = 0.f;
    float4 acc = make_float4(0.f, 0.f, 0.f, 0.f);
    for (int s = 0; s <= t; ++s) {
        const float4 kf = *(const float4*)(k + (size_t)s * (NKVH * DH) + kvh * DH + lane * 4);
        float d = qv.x * kf.x + qv.y * kf.y + qv.z * kf.z + qv.w * kf.w;
        #pragma unroll
        for (int off2 = 16; off2 > 0; off2 >>= 1) d += __shfl_xor(d, off2, 32);
        const float mn = fmaxf(m, d);
        const float sc = __expf(m - mn);
        const float p  = __expf(d - mn);
        const float4 vf = *(const float4*)(v + (size_t)s * (NKVH * DH) + kvh * DH + lane * 4);
        acc.x = acc.x * sc + p * vf.x;
        acc.y = acc.y * sc + p * vf.y;
        acc.z = acc.z * sc + p * vf.z;
        acc.w = acc.w * sc + p * vf.w;
        l = l * sc + p;
        m = mn;
    }
    const float inv = 1.0f / l;
    float4 r = make_float4(acc.x * inv, acc.y * inv, acc.z * inv, acc.w * inv);
    *(float4*)(o + (size_t)t * (NHQ * DH) + head * DH + lane * 4) = r;
}

// =====================================================================
// Router: softmax over 16 experts, top-2, renormalize
// =====================================================================
__global__ __launch_bounds__(256) void router_kernel(
    const float* __restrict__ y, const float* __restrict__ gw,
    int* __restrict__ tkid, float* __restrict__ tkw)
{
    __shared__ float ys[HH];
    __shared__ float lg[NE];
    const int t = blockIdx.x;
    for (int i = threadIdx.x; i < HH; i += 256) ys[i] = y[(size_t)t * HH + i];
    __syncthreads();
    if (threadIdx.x < NE) {
        float acc = 0.f;
        for (int h = 0; h < HH; ++h) acc += ys[h] * gw[h * NE + threadIdx.x];
        lg[threadIdx.x] = acc;
    }
    __syncthreads();
    if (threadIdx.x == 0) {
        float mx = lg[0];
        for (int e = 1; e < NE; ++e) mx = fmaxf(mx, lg[e]);
        float p[NE]; float sum = 0.f;
        for (int e = 0; e < NE; ++e) { p[e] = __expf(lg[e] - mx); sum += p[e]; }
        const float inv = 1.0f / sum;
        for (int e = 0; e < NE; ++e) p[e] *= inv;
        int i0 = 0; float v0 = p[0];
        for (int e = 1; e < NE; ++e) if (p[e] > v0) { v0 = p[e]; i0 = e; }
        int i1 = -1; float v1 = -1.f;
        for (int e = 0; e < NE; ++e) if (e != i0 && p[e] > v1) { v1 = p[e]; i1 = e; }
        const float rn = 1.0f / (v0 + v1);
        tkid[t * 2 + 0] = i0; tkw[t * 2 + 0] = v0 * rn;
        tkid[t * 2 + 1] = i1; tkw[t * 2 + 1] = v1 * rn;
    }
}

// ----- routing bookkeeping -----
__global__ void zero_kernel(int* counts, int* fill)
{
    if (threadIdx.x < NE) { counts[threadIdx.x] = 0; fill[threadIdx.x] = 0; }
}
__global__ __launch_bounds__(256) void count_kernel(
    const int* __restrict__ tkid, int* __restrict__ counts)
{
    const int gid = blockIdx.x * 256 + threadIdx.x;
    if (gid < NSLOT) atomicAdd(&counts[tkid[gid]], 1);
}
__global__ void scan_kernel(const int* __restrict__ counts, int* __restrict__ offs)
{
    if (threadIdx.x == 0) {
        offs[0] = 0;
        for (int e = 0; e < NE; ++e) offs[e + 1] = offs[e] + counts[e];
    }
}
__global__ __launch_bounds__(256) void scatter_kernel(
    const int* __restrict__ tkid, const float* __restrict__ tkw,
    const int* __restrict__ offs, int* __restrict__ fill,
    int* __restrict__ slot_tok, float* __restrict__ slotw,
    int* __restrict__ slot_of)
{
    const int gid = blockIdx.x * 256 + threadIdx.x;
    if (gid >= NSLOT) return;
    const int e = tkid[gid];
    const int pos = atomicAdd(&fill[e], 1);
    const int slot = offs[e] + pos;
    slot_tok[slot] = gid >> 1;
    slotw[slot]    = tkw[gid];
    slot_of[gid]   = slot;
}

// =====================================================================
// MoE fused gate+up GEMM (gathered rows) with u*silu(g) epilogue
// grid: (rowTiles, I/128, E); inter[slot, I]
// =====================================================================
__global__ __launch_bounds__(256) void moe_gateup_kernel(
    const float* __restrict__ Y, const float* __restrict__ Wg,
    const float* __restrict__ Wu, const int* __restrict__ offs,
    const int* __restrict__ slot_tok, float* __restrict__ inter)
{
    const int e   = blockIdx.z;
    const int off = offs[e];
    const int cnt = offs[e + 1] - off;
    const int bm  = blockIdx.x * 32;
    if (bm >= cnt) return;
    const int bn  = blockIdx.y * 128;
    __shared__ _Float16 As[32][LSTRIDE];
    __shared__ _Float16 BgT[128][LSTRIDE];
    __shared__ _Float16 BuT[128][LSTRIDE];
    const int tid = threadIdx.x;
    const int wave = tid >> 5, lane = tid & 31;
    const int tm = (wave >> 2) << 4;
    const int tn = (wave & 3) << 5;
    const int lr = tid >> 3;
    const int la = (tid & 7) << 2;
    const int kk  = (tid >> 4) << 1;
    const int nn0 = (tid & 15) << 3;
    const int hhalf = lane >> 4, mr = lane & 15;
    const float* wg = Wg + (size_t)e * HH * II;
    const float* wu = Wu + (size_t)e * HH * II;
    const int grow = bm + lr;
    const int tok  = (grow < cnt) ? slot_tok[off + grow] : -1;
    v8f ag0 = {}, ag1 = {}, au0 = {}, au1 = {};
    for (int k0 = 0; k0 < HH; k0 += 32) {
        {
            v4h a;
            if (tok >= 0) {
                const float4 f = *(const float4*)(Y + (size_t)tok * HH + k0 + la);
                a[0] = (_Float16)f.x; a[1] = (_Float16)f.y;
                a[2] = (_Float16)f.z; a[3] = (_Float16)f.w;
            } else {
                a[0] = a[1] = a[2] = a[3] = (_Float16)0.f;
            }
            *(v4h*)&As[lr][la] = a;
        }
        {
            const size_t boff = (size_t)(k0 + kk) * II + bn + nn0;
            const float4 g00 = *(const float4*)(wg + boff);
            const float4 g01 = *(const float4*)(wg + boff + 4);
            const float4 g10 = *(const float4*)(wg + boff + II);
            const float4 g11 = *(const float4*)(wg + boff + II + 4);
            const float4 u00 = *(const float4*)(wu + boff);
            const float4 u01 = *(const float4*)(wu + boff + 4);
            const float4 u10 = *(const float4*)(wu + boff + II);
            const float4 u11 = *(const float4*)(wu + boff + II + 4);
            v2h t;
            t[0] = (_Float16)g00.x; t[1] = (_Float16)g10.x; *(v2h*)&BgT[nn0+0][kk] = t;
            t[0] = (_Float16)g00.y; t[1] = (_Float16)g10.y; *(v2h*)&BgT[nn0+1][kk] = t;
            t[0] = (_Float16)g00.z; t[1] = (_Float16)g10.z; *(v2h*)&BgT[nn0+2][kk] = t;
            t[0] = (_Float16)g00.w; t[1] = (_Float16)g10.w; *(v2h*)&BgT[nn0+3][kk] = t;
            t[0] = (_Float16)g01.x; t[1] = (_Float16)g11.x; *(v2h*)&BgT[nn0+4][kk] = t;
            t[0] = (_Float16)g01.y; t[1] = (_Float16)g11.y; *(v2h*)&BgT[nn0+5][kk] = t;
            t[0] = (_Float16)g01.z; t[1] = (_Float16)g11.z; *(v2h*)&BgT[nn0+6][kk] = t;
            t[0] = (_Float16)g01.w; t[1] = (_Float16)g11.w; *(v2h*)&BgT[nn0+7][kk] = t;
            t[0] = (_Float16)u00.x; t[1] = (_Float16)u10.x; *(v2h*)&BuT[nn0+0][kk] = t;
            t[0] = (_Float16)u00.y; t[1] = (_Float16)u10.y; *(v2h*)&BuT[nn0+1][kk] = t;
            t[0] = (_Float16)u00.z; t[1] = (_Float16)u10.z; *(v2h*)&BuT[nn0+2][kk] = t;
            t[0] = (_Float16)u00.w; t[1] = (_Float16)u10.w; *(v2h*)&BuT[nn0+3][kk] = t;
            t[0] = (_Float16)u01.x; t[1] = (_Float16)u11.x; *(v2h*)&BuT[nn0+4][kk] = t;
            t[0] = (_Float16)u01.y; t[1] = (_Float16)u11.y; *(v2h*)&BuT[nn0+5][kk] = t;
            t[0] = (_Float16)u01.z; t[1] = (_Float16)u11.z; *(v2h*)&BuT[nn0+6][kk] = t;
            t[0] = (_Float16)u01.w; t[1] = (_Float16)u11.w; *(v2h*)&BuT[nn0+7][kk] = t;
        }
        if (k0 + 32 < HH) {
            __builtin_prefetch(wg + (size_t)(k0 + 32 + kk) * II + bn + nn0, 0, 1);
            __builtin_prefetch(wu + (size_t)(k0 + 32 + kk) * II + bn + nn0, 0, 1);
        }
        __syncthreads();
        const v16h af  = ldfrag(&As[tm + mr][0], hhalf);
        const v16h bg0 = ldfrag(&BgT[tn + mr][0], hhalf);
        const v16h bg1 = ldfrag(&BgT[tn + 16 + mr][0], hhalf);
        const v16h bu0 = ldfrag(&BuT[tn + mr][0], hhalf);
        const v16h bu1 = ldfrag(&BuT[tn + 16 + mr][0], hhalf);
        ag0 = __builtin_amdgcn_wmma_f32_16x16x32_f16(false, af, false, bg0, (short)0, ag0, false, false);
        ag1 = __builtin_amdgcn_wmma_f32_16x16x32_f16(false, af, false, bg1, (short)0, ag1, false, false);
        au0 = __builtin_amdgcn_wmma_f32_16x16x32_f16(false, af, false, bu0, (short)0, au0, false, false);
        au1 = __builtin_amdgcn_wmma_f32_16x16x32_f16(false, af, false, bu1, (short)0, au1, false, false);
        __syncthreads();
    }
    const int col0 = bn + tn + mr;
    const int col1 = col0 + 16;
    #pragma unroll
    for (int r = 0; r < 8; ++r) {
        const int row = bm + tm + (hhalf << 3) + r;
        if (row < cnt) {
            const float g0 = ag0[r], u0 = au0[r];
            const float g1 = ag1[r], u1 = au1[r];
            inter[(size_t)(off + row) * II + col0] = u0 * g0 / (1.0f + __expf(-g0));
            inter[(size_t)(off + row) * II + col1] = u1 * g1 / (1.0f + __expf(-g1));
        }
    }
}

// =====================================================================
// MoE down GEMM: dbuf[slot, H] = inter[slot, I] @ w_down[e]
// rows per expert are contiguous in slot space -> plain GEMM per expert
// =====================================================================
__global__ __launch_bounds__(256) void moe_down_kernel(
    const float* __restrict__ inter, const float* __restrict__ Wd,
    const int* __restrict__ offs, float* __restrict__ dbuf)
{
    const int e   = blockIdx.z;
    const int off = offs[e];
    const int cnt = offs[e + 1] - off;
    const int bm  = blockIdx.x * 32;
    if (bm >= cnt) return;
    const int bn  = blockIdx.y * 128;
    __shared__ _Float16 As[32][LSTRIDE];
    __shared__ _Float16 BsT[128][LSTRIDE];
    const int tid = threadIdx.x;
    const int wave = tid >> 5, lane = tid & 31;
    const int tm = (wave >> 2) << 4;
    const int tn = (wave & 3) << 5;
    const int lr = tid >> 3;
    const int la = (tid & 7) << 2;
    const int kk  = (tid >> 4) << 1;
    const int nn0 = (tid & 15) << 3;
    const int hhalf = lane >> 4, mr = lane & 15;
    const float* A = inter + (size_t)off * II;
    const float* B = Wd + (size_t)e * II * HH;
    v8f acc0 = {}, acc1 = {};
    for (int k0 = 0; k0 < II; k0 += 32) {
        {
            const int row = bm + lr;
            v4h a;
            if (row < cnt) {
                const float4 f = *(const float4*)(A + (size_t)row * II + k0 + la);
                a[0] = (_Float16)f.x; a[1] = (_Float16)f.y;
                a[2] = (_Float16)f.z; a[3] = (_Float16)f.w;
            } else {
                a[0] = a[1] = a[2] = a[3] = (_Float16)0.f;
            }
            *(v4h*)&As[lr][la] = a;
        }
        {
            const float* bp0 = B + (size_t)(k0 + kk) * HH + bn + nn0;
            const float* bp1 = bp0 + HH;
            const float4 r00 = *(const float4*)(bp0);
            const float4 r01 = *(const float4*)(bp0 + 4);
            const float4 r10 = *(const float4*)(bp1);
            const float4 r11 = *(const float4*)(bp1 + 4);
            v2h t;
            t[0] = (_Float16)r00.x; t[1] = (_Float16)r10.x; *(v2h*)&BsT[nn0+0][kk] = t;
            t[0] = (_Float16)r00.y; t[1] = (_Float16)r10.y; *(v2h*)&BsT[nn0+1][kk] = t;
            t[0] = (_Float16)r00.z; t[1] = (_Float16)r10.z; *(v2h*)&BsT[nn0+2][kk] = t;
            t[0] = (_Float16)r00.w; t[1] = (_Float16)r10.w; *(v2h*)&BsT[nn0+3][kk] = t;
            t[0] = (_Float16)r01.x; t[1] = (_Float16)r11.x; *(v2h*)&BsT[nn0+4][kk] = t;
            t[0] = (_Float16)r01.y; t[1] = (_Float16)r11.y; *(v2h*)&BsT[nn0+5][kk] = t;
            t[0] = (_Float16)r01.z; t[1] = (_Float16)r11.z; *(v2h*)&BsT[nn0+6][kk] = t;
            t[0] = (_Float16)r01.w; t[1] = (_Float16)r11.w; *(v2h*)&BsT[nn0+7][kk] = t;
        }
        if (k0 + 32 < II)
            __builtin_prefetch(B + (size_t)(k0 + 32 + kk) * HH + bn + nn0, 0, 1);
        __syncthreads();
        const v16h af  = ldfrag(&As[tm + mr][0], hhalf);
        const v16h bf0 = ldfrag(&BsT[tn + mr][0], hhalf);
        const v16h bf1 = ldfrag(&BsT[tn + 16 + mr][0], hhalf);
        acc0 = __builtin_amdgcn_wmma_f32_16x16x32_f16(false, af, false, bf0, (short)0, acc0, false, false);
        acc1 = __builtin_amdgcn_wmma_f32_16x16x32_f16(false, af, false, bf1, (short)0, acc1, false, false);
        __syncthreads();
    }
    const int col0 = bn + tn + mr;
    const int col1 = col0 + 16;
    #pragma unroll
    for (int r = 0; r < 8; ++r) {
        const int row = bm + tm + (hhalf << 3) + r;
        if (row < cnt) {
            dbuf[(size_t)(off + row) * HH + col0] = acc0[r];
            dbuf[(size_t)(off + row) * HH + col1] = acc1[r];
        }
    }
}

// =====================================================================
// Combine: out[t] += w0 * dbuf[slot(t,0)] + w1 * dbuf[slot(t,1)]
// deterministic (fixed k order; no atomics on data)
// =====================================================================
__global__ __launch_bounds__(256) void combine_kernel(
    const float* __restrict__ dbuf, const int* __restrict__ slot_of,
    const float* __restrict__ slotw, float* __restrict__ out)
{
    const int gid = blockIdx.x * 256 + threadIdx.x;
    if (gid >= TT * (HH / 4)) return;
    const int t = gid / (HH / 4);
    const int c = (gid % (HH / 4)) * 4;
    const int s0 = slot_of[t * 2 + 0];
    const int s1 = slot_of[t * 2 + 1];
    const float w0 = slotw[s0], w1 = slotw[s1];
    float4 o = *(float4*)(out + (size_t)t * HH + c);
    const float4 a = *(const float4*)(dbuf + (size_t)s0 * HH + c);
    const float4 b = *(const float4*)(dbuf + (size_t)s1 * HH + c);
    o.x += w0 * a.x + w1 * b.x;
    o.y += w0 * a.y + w1 * b.y;
    o.z += w0 * a.z + w1 * b.z;
    o.w += w0 * a.w + w1 * b.w;
    *(float4*)(out + (size_t)t * HH + c) = o;
}

// =====================================================================
// Host-side orchestration
// =====================================================================
extern "C" void kernel_launch(void* const* d_in, const int* in_sizes, int n_in,
                              void* d_out, int out_size, void* d_ws, size_t ws_size,
                              hipStream_t stream)
{
    const float* hs     = (const float*)d_in[0];
    const int*   pos    = (const int*)  d_in[1];
    const float* ln1w   = (const float*)d_in[2];
    const float* ln2w   = (const float*)d_in[3];
    const float* wq     = (const float*)d_in[4];
    const float* bq     = (const float*)d_in[5];
    const float* wk     = (const float*)d_in[6];
    const float* bk     = (const float*)d_in[7];
    const float* wv     = (const float*)d_in[8];
    const float* bv     = (const float*)d_in[9];
    const float* wo     = (const float*)d_in[10];
    const float* gate_w = (const float*)d_in[11];
    const float* w_gate = (const float*)d_in[12];
    const float* w_up   = (const float*)d_in[13];
    const float* w_down = (const float*)d_in[14];
    float* out = (float*)d_out;

    // ---- workspace layout ----
    float* fws = (float*)d_ws;
    size_t o = 0;
    float* xn    = fws + o; o += (size_t)TT * HH;
    float* qb    = fws + o; o += (size_t)TT * NHQ  * DH;
    float* kb    = fws + o; o += (size_t)TT * NKVH * DH;
    float* vb    = fws + o; o += (size_t)TT * NKVH * DH;
    float* ao    = fws + o; o += (size_t)TT * NHQ  * DH;
    float* yb    = fws + o; o += (size_t)TT * HH;
    float* inter = fws + o; o += (size_t)NSLOT * II;
    float* dbuf  = fws + o; o += (size_t)NSLOT * HH;
    float* tkw   = fws + o; o += NSLOT;
    float* slotw = fws + o; o += NSLOT;
    int* ib       = (int*)(fws + o);
    int* tkid     = ib; ib += NSLOT;
    int* counts   = ib; ib += NE;
    int* offs     = ib; ib += NE + 1;
    int* fill     = ib; ib += NE;
    int* slot_tok = ib; ib += NSLOT;
    int* slot_of  = ib; ib += NSLOT;

    // ---- attention block ----
    rmsnorm_kernel<<<TT, 256, 0, stream>>>(hs, ln1w, xn);
    gemm_wmma_kernel<<<dim3(TT/32, (NHQ*DH)/128), 256, 0, stream>>>(
        xn, wq, bq, nullptr, qb, TT, NHQ*DH, HH);
    gemm_wmma_kernel<<<dim3(TT/32, (NKVH*DH)/128), 256, 0, stream>>>(
        xn, wk, bk, nullptr, kb, TT, NKVH*DH, HH);
    gemm_wmma_kernel<<<dim3(TT/32, (NKVH*DH)/128), 256, 0, stream>>>(
        xn, wv, bv, nullptr, vb, TT, NKVH*DH, HH);
    rope_kernel<<<(TT*NHQ*64 + 255)/256, 256, 0, stream>>>(qb, pos, NHQ);
    rope_kernel<<<(TT*NKVH*64 + 255)/256, 256, 0, stream>>>(kb, pos, NKVH);
    attn_kernel<<<(TT*NHQ)/8, 256, 0, stream>>>(qb, kb, vb, ao);
    gemm_wmma_kernel<<<dim3(TT/32, HH/128), 256, 0, stream>>>(
        ao, wo, nullptr, hs, out, TT, HH, NHQ*DH);   // out = hs + attn @ wo

    // ---- MoE block ----
    rmsnorm_kernel<<<TT, 256, 0, stream>>>(out, ln2w, yb);
    router_kernel<<<TT, 256, 0, stream>>>(yb, gate_w, tkid, tkw);
    zero_kernel<<<1, 32, 0, stream>>>(counts, fill);
    count_kernel<<<(NSLOT + 255)/256, 256, 0, stream>>>(tkid, counts);
    scan_kernel<<<1, 1, 0, stream>>>(counts, offs);
    scatter_kernel<<<(NSLOT + 255)/256, 256, 0, stream>>>(
        tkid, tkw, offs, fill, slot_tok, slotw, slot_of);
    moe_gateup_kernel<<<dim3(TT/32, II/128, NE), 256, 0, stream>>>(
        yb, w_gate, w_up, offs, slot_tok, inter);
    moe_down_kernel<<<dim3(TT/32, HH/128, NE), 256, 0, stream>>>(
        inter, w_down, offs, dbuf);
    combine_kernel<<<(TT*(HH/4) + 255)/256, 256, 0, stream>>>(
        dbuf, slot_of, slotw, out);
}
